// Encoder_29892972380736
// MI455X (gfx1250) — matrix operand order repaired and checked by reading, not compiled, not run on hardware
//
#include <hip/hip_runtime.h>
#include <hip/hip_bf16.h>

typedef __attribute__((ext_vector_type(16))) __bf16 v16bf;
typedef __attribute__((ext_vector_type(8)))  float  v8f;

#define GDIM 128
#define LDS_STRIDE 136   // 128 + 8 bf16 pad: lane m -> bank 4m % 64, conflict-free

// ---------------------------------------------------------------- utilities
__global__ __launch_bounds__(256)
void k_zero(float* __restrict__ p, int n) {
    int i = blockIdx.x * 256 + threadIdx.x;
    if (i < n) p[i] = 0.0f;
}

__global__ __launch_bounds__(256)
void k_fill1(float* __restrict__ p, int n) {
    int i = blockIdx.x * 256 + threadIdx.x;
    if (i < n) p[i] = 1.0f;   // self-loop contributes 1 to every node's degree
}

__global__ __launch_bounds__(256)
void k_deg_accum(const long long* __restrict__ ei, float* __restrict__ deg, int E) {
    int e = blockIdx.x * 256 + threadIdx.x;
    if (e < E) {
        int d = (int)ei[(size_t)E + e];   // dst row of edge_index
        __hip_atomic_fetch_add(&deg[d], 1.0f, __ATOMIC_RELAXED, __HIP_MEMORY_SCOPE_AGENT);
    }
}

__global__ __launch_bounds__(256)
void k_dinv(const float* __restrict__ deg, float* __restrict__ dinv, int n) {
    int i = blockIdx.x * 256 + threadIdx.x;
    if (i < n) dinv[i] = rsqrtf(deg[i]);
}

// ------------------------------------------------------- WMMA GEMM: t = h @ W
// One block = 128 threads = 4 waves; each wave computes a 16x128 output stripe
// as 8 (16x16) C tiles, each accumulated over 4 k-blocks of K=32 (DIM=128).
__global__ __launch_bounds__(128)
void k_gemm_wmma(const float* __restrict__ hin,   // [N,128] f32 row-major
                 const float* __restrict__ W,     // [128,128] f32 row-major (k,n)
                 float* __restrict__ tout,        // [N,128] f32
                 int nRows)
{
    __shared__ __bf16 Wlds[GDIM * LDS_STRIDE];

    const int tid = threadIdx.x;
    // Stage W into LDS as bf16 (32 KB of the 320 KB WGP LDS).
    for (int i = tid; i < GDIM * GDIM; i += 128) {
        int k = i >> 7;
        int n = i & (GDIM - 1);
        Wlds[k * LDS_STRIDE + n] = (__bf16)W[i];
    }
    __syncthreads();

    const int wave    = tid >> 5;
    const int lane    = tid & 31;
    const int half    = lane >> 4;   // 0: K 0-7/16-23 side, 1: K 8-15/24-31 side
    const int m       = lane & 15;
    const int rowBase = (blockIdx.x * 4 + wave) * 16;
    if (rowBase >= nRows) return;    // wave-uniform exit (N=50000 is 16-aligned)

    // A fragments: 16x32 bf16 per k-block, per ISA 16-bit A layout.
    v16bf afrag[4];
    const float* hrow = hin + (size_t)(rowBase + m) * GDIM;
    #pragma unroll
    for (int kb = 0; kb < 4; ++kb) {
        const int k0 = kb * 32 + half * 8;
        float4 lo0 = *(const float4*)(hrow + k0);
        float4 lo1 = *(const float4*)(hrow + k0 + 4);
        float4 hi0 = *(const float4*)(hrow + k0 + 16);
        float4 hi1 = *(const float4*)(hrow + k0 + 20);
        afrag[kb][0]  = (__bf16)lo0.x; afrag[kb][1]  = (__bf16)lo0.y;
        afrag[kb][2]  = (__bf16)lo0.z; afrag[kb][3]  = (__bf16)lo0.w;
        afrag[kb][4]  = (__bf16)lo1.x; afrag[kb][5]  = (__bf16)lo1.y;
        afrag[kb][6]  = (__bf16)lo1.z; afrag[kb][7]  = (__bf16)lo1.w;
        afrag[kb][8]  = (__bf16)hi0.x; afrag[kb][9]  = (__bf16)hi0.y;
        afrag[kb][10] = (__bf16)hi0.z; afrag[kb][11] = (__bf16)hi0.w;
        afrag[kb][12] = (__bf16)hi1.x; afrag[kb][13] = (__bf16)hi1.y;
        afrag[kb][14] = (__bf16)hi1.z; afrag[kb][15] = (__bf16)hi1.w;
    }

    #pragma unroll
    for (int nt = 0; nt < 8; ++nt) {
        v8f c = {};
        #pragma unroll
        for (int kb = 0; kb < 4; ++kb) {
            // B 32x16 bf16: lane = column n (m), elements = K 8h+0..7 / 16+8h+0..7
            const int kRow = kb * 32 + half * 8;
            union { uint4 u[2]; v16bf v; } b;
            const __bf16* bp = &Wlds[kRow * LDS_STRIDE + nt * 16 + m];
            #pragma unroll
            for (int s = 0; s < 8; ++s) {
                b.v[s]     = bp[s * LDS_STRIDE];
                b.v[s + 8] = bp[(s + 16) * LDS_STRIDE];
            }
            c = __builtin_amdgcn_wmma_f32_16x16x32_bf16(
                    /*neg_a=*/false, afrag[kb], /*neg_b=*/false, b.v,
                    /*c_mod=*/(short)0, c, /*reuse_a=*/false, /*reuse_b=*/false);
        }
        // D 16x16 f32: VGPR r -> rows rowBase+r (lanes 0-15) / rowBase+r+8 (16-31)
        float* tb = tout + (size_t)(rowBase + half * 8) * GDIM + nt * 16 + m;
        #pragma unroll
        for (int r = 0; r < 8; ++r)
            tb[(size_t)r * GDIM] = c[r];
    }
}

// -------------------------------------------- self-loop init: agg = t * dinv^2
__global__ __launch_bounds__(256)
void k_selfloop(const float* __restrict__ t, const float* __restrict__ dinv,
                float* __restrict__ agg, int total) {
    int i = blockIdx.x * 256 + threadIdx.x;
    if (i < total) {
        int n = i >> 7;
        float di = dinv[n];
        agg[i] = t[i] * di * di;
    }
}

// ------------------------------------- edge scatter: agg[dst] += norm * t[src]
// One wave per edge; each lane handles 4 contiguous floats (float4 gather,
// 4 native f32 atomics into L2-resident agg).
__global__ __launch_bounds__(256)
void k_scatter(const long long* __restrict__ ei, const float* __restrict__ t,
               const float* __restrict__ dinv, float* __restrict__ agg, int E) {
    const int lane = threadIdx.x & 31;
    const int wid  = blockIdx.x * 8 + (threadIdx.x >> 5);
    const int nw   = gridDim.x * 8;
    for (int e = wid; e < E; e += nw) {
        const int s = (int)ei[e];
        const int d = (int)ei[(size_t)E + e];
        if (e + nw < E) __builtin_prefetch(&ei[e + nw], 0, 0);  // global_prefetch_b8
        const float nrm = dinv[s] * dinv[d];
        const float4 v = *(const float4*)(t + (size_t)s * GDIM + lane * 4);
        float* ap = agg + (size_t)d * GDIM + lane * 4;
        __hip_atomic_fetch_add(ap + 0, v.x * nrm, __ATOMIC_RELAXED, __HIP_MEMORY_SCOPE_AGENT);
        __hip_atomic_fetch_add(ap + 1, v.y * nrm, __ATOMIC_RELAXED, __HIP_MEMORY_SCOPE_AGENT);
        __hip_atomic_fetch_add(ap + 2, v.z * nrm, __ATOMIC_RELAXED, __HIP_MEMORY_SCOPE_AGENT);
        __hip_atomic_fetch_add(ap + 3, v.w * nrm, __ATOMIC_RELAXED, __HIP_MEMORY_SCOPE_AGENT);
    }
}

// ---------------------- fused bias + ReLU + global_add_pool (out[g, li*128+c])
__global__ __launch_bounds__(256)
void k_bias_relu_pool(const float* __restrict__ agg, const float* __restrict__ b,
                      const long long* __restrict__ batch,
                      float* __restrict__ h, float* __restrict__ out,
                      int total, int li, int poolStride) {
    int i = blockIdx.x * 256 + threadIdx.x;
    if (i < total) {
        int n = i >> 7;
        int c = i & (GDIM - 1);
        float v = agg[i] + b[c];
        v = v > 0.0f ? v : 0.0f;
        h[i] = v;
        int g = (int)batch[n];
        __hip_atomic_fetch_add(out + (size_t)g * poolStride + li * GDIM + c, v,
                               __ATOMIC_RELAXED, __HIP_MEMORY_SCOPE_AGENT);
    }
}

// ---------------------------------------------------------------------- host
extern "C" void kernel_launch(void* const* d_in, const int* in_sizes, int n_in,
                              void* d_out, int out_size, void* d_ws, size_t ws_size,
                              hipStream_t stream) {
    const float*     x     = (const float*)d_in[0];
    const long long* ei    = (const long long*)d_in[1];
    const long long* batch = (const long long*)d_in[2];
    const float*     Ws    = (const float*)d_in[3];
    const float*     bs    = (const float*)d_in[4];
    float*           out   = (float*)d_out;

    const int N = in_sizes[2];            // 50000
    const int E = in_sizes[1] / 2;        // 600000
    const int L = in_sizes[3] / (GDIM * GDIM);  // 5
    const int poolStride = L * GDIM;      // 640
    const int total = N * GDIM;

    // Workspace layout (f32), 256B-aligned slabs.
    float* ws   = (float*)d_ws;
    float* dinv = ws;                              // N (padded to 50048)
    float* t    = dinv + ((N + 63) & ~63);
    float* agg  = t + (size_t)total;
    float* h    = agg + (size_t)total;

    const int g256   = (total + 255) / 256;
    const int gN     = (N + 255) / 256;
    const int gE     = (E + 255) / 256;
    const int gOut   = (out_size + 255) / 256;
    const int gGemm  = (N + 63) / 64;              // 4 waves x 16 rows per block

    // Output accumulates pooled sums -> must start at zero (ws/out are poisoned).
    k_zero<<<gOut, 256, 0, stream>>>(out, out_size);

    // Degree (with self-loop) -> dinv = deg^-1/2. Reuse `t` as deg scratch.
    k_fill1<<<gN, 256, 0, stream>>>(t, N);
    k_deg_accum<<<gE, 256, 0, stream>>>(ei, t, E);
    k_dinv<<<gN, 256, 0, stream>>>(t, dinv, N);

    const float* hcur = x;
    for (int li = 0; li < L; ++li) {
        k_gemm_wmma<<<gGemm, 128, 0, stream>>>(hcur, Ws + (size_t)li * GDIM * GDIM, t, N);
        k_selfloop<<<g256, 256, 0, stream>>>(t, dinv, agg, total);
        k_scatter<<<1024, 256, 0, stream>>>(ei, t, dinv, agg, E);
        k_bias_relu_pool<<<g256, 256, 0, stream>>>(agg, bs + (size_t)li * GDIM, batch,
                                                   h, out, total, li, poolStride);
        hcur = h;
    }
}
